// HSDM_32014686224786
// MI455X (gfx1250) — matrix operand order repaired and checked by reading, not compiled, not run on hardware
//
#include <hip/hip_runtime.h>

#define BN    3888
#define JJ    17
#define CC    512
#define HH    8
#define DDIM  64
#define NC3   1536

// ---------- LDS layout (bytes), all offsets 16B aligned ----------
#define OFF_AT   0          // float[8*17*20]        = 10880  attention scores/probs (f32)
#define OFF_RED  10880      // float[36]             = 144    LN sum / sumsq per row
#define OFF_XB   11024      // u16[32*520]           = 33280  x in bf16 (32-row tile, rows>=17 zeroed)
#define OFF_Q    44304      // u16[8*17*72]          = 19584  q[h][j][d]   (q+k region reused as xa)
#define OFF_K    63888      // u16[8*17*72]          = 19584  k[h][j][d]
#define OFF_VT   83472      // u16[8*64*40]          = 40960  v transposed [h][d][j]
#define OFF_AP   124432     // u16[8*17*40]          = 10880  probs bf16, K-padded to 32
#define SMEM_BYTES 135312
// xa tile (u16[32*520] = 33280 B) aliases [OFF_Q .. OFF_Q+39168) -- q and k are dead by then.

typedef __attribute__((ext_vector_type(16))) __bf16        bf16x16;
typedef __attribute__((ext_vector_type(8)))  float         f32x8;
typedef __attribute__((ext_vector_type(4)))  unsigned int  u32x4;

union Frag { bf16x16 v; u32x4 u[2]; };

__device__ __forceinline__ unsigned short f2bf(float f) {
    union { float f; unsigned u; } a; a.f = f;
    unsigned r = a.u + 0x7FFFu + ((a.u >> 16) & 1u);   // round-to-nearest-even
    return (unsigned short)(r >> 16);
}
__device__ __forceinline__ float bf2f(unsigned short h) {
    union { float f; unsigned u; } a; a.u = ((unsigned)h) << 16;
    return a.f;
}

// A fragment (16x32, bf16) from LDS, row-major [m][k], row stride ld (elems).
// ISA 7.12.2: lanes 0-15 hold row m=lane, K {0..7,16..23}; lanes 16-31 K {8..15,24..31}.
// FAST variant: tile is fully backed (padding rows pre-zeroed in LDS) -> no selects.
__device__ __forceinline__ Frag fragA_fast(const unsigned short* p0, int ld,
                                           int row0, int koff) {
    const int lane = threadIdx.x & 31;
    const unsigned short* p = p0 + (row0 + (lane & 15)) * ld + koff + ((lane >> 4) << 3);
    Frag f;
    f.u[0] = *(const u32x4*)(p);                 // K base .. +7   (ds_load_b128)
    f.u[1] = *(const u32x4*)(p + 16);            // K base+16 .. +23
    return f;
}

// Checked variant (rows >= valid read as zeros) for small attention tiles.
__device__ __forceinline__ Frag fragA_lds(const unsigned short* p0, int ld,
                                          int row0, int valid, int koff) {
    const int lane = threadIdx.x & 31;
    const int r    = row0 + (lane & 15);
    const bool ok  = (r < valid);
    const unsigned short* p = p0 + (ok ? r : 0) * ld + koff + ((lane >> 4) << 3);
    Frag f;
    f.u[0] = *(const u32x4*)(p);
    f.u[1] = *(const u32x4*)(p + 16);
    if (!ok) { u32x4 z = {0u,0u,0u,0u}; f.u[0] = z; f.u[1] = z; }
    return f;
}

// B fragment (32x16, bf16) from LDS stored as [n][k] (k contiguous), row stride ld.
// Lane n holds column n; lanes 0-15: K {0..7,16..23}, lanes 16-31: K {8..15,24..31} (matches A).
__device__ __forceinline__ Frag fragB_lds(const unsigned short* p0, int ld,
                                          int col0, int valid, int koff) {
    const int lane = threadIdx.x & 31;
    const int c    = col0 + (lane & 15);
    const bool ok  = (c < valid);
    const unsigned short* p = p0 + (ok ? c : 0) * ld + koff + ((lane >> 4) << 3);
    Frag f;
    f.u[0] = *(const u32x4*)(p);
    f.u[1] = *(const u32x4*)(p + 16);
    if (!ok) { u32x4 z = {0u,0u,0u,0u}; f.u[0] = z; f.u[1] = z; }
    return f;
}

// Same B fragment but straight from global (L2-resident transposed weights).
__device__ __forceinline__ Frag fragB_glb(const unsigned short* __restrict__ g, int ld,
                                          int col0, int koff) {
    const int lane = threadIdx.x & 31;
    const unsigned short* p = g + (size_t)(col0 + (lane & 15)) * ld
                                + koff + ((lane >> 4) << 3);
    Frag f;
    f.u[0] = *(const u32x4*)(p);                 // global_load_b128
    f.u[1] = *(const u32x4*)(p + 16);
    return f;
}

__device__ __forceinline__ f32x8 wmma_bf16(const Frag& a, const Frag& b, f32x8 c) {
    return __builtin_amdgcn_wmma_f32_16x16x32_bf16(false, a.v, false, b.v,
                                                   (short)0, c, false, false);
}

// ---------------- weight convert + transpose (fp32 -> bf16, [k][n] -> [n][k]) -------------
__global__ __launch_bounds__(256)
void cvt_weights(const float* __restrict__ Wqkv, const float* __restrict__ Wproj,
                 unsigned short* __restrict__ WqkvT, unsigned short* __restrict__ WprojT) {
    const int i = blockIdx.x * 256 + threadIdx.x;
    if (i < CC * NC3) {
        const int k = i / NC3, n = i % NC3;
        WqkvT[(size_t)n * CC + k] = f2bf(Wqkv[i]);
    }
    if (i < CC * CC) {
        const int k = i / CC, n = i % CC;
        WprojT[(size_t)n * CC + k] = f2bf(Wproj[i]);
    }
}

// ---------------- fused HSDM: one workgroup (8 wave32) per frame ----------------
__global__ __launch_bounds__(256)
void hsdm_fused(const float* __restrict__ x,
                const float* __restrict__ bqkv,
                const float* __restrict__ bproj,
                const unsigned short* __restrict__ WqkvT,
                const unsigned short* __restrict__ WprojT,
                float* __restrict__ out) {
    extern __shared__ char smem[];
    float*          sAT  = (float*)(smem + OFF_AT);
    float*          sRED = (float*)(smem + OFF_RED);
    unsigned short* sXB  = (unsigned short*)(smem + OFF_XB);
    unsigned short* sQ   = (unsigned short*)(smem + OFF_Q);
    unsigned short* sK   = (unsigned short*)(smem + OFF_K);
    unsigned short* sVT  = (unsigned short*)(smem + OFF_VT);
    unsigned short* sAP  = (unsigned short*)(smem + OFF_AP);
    unsigned short* sXA  = sQ;   // q+k region is dead once QK^T is done; reuse for xa

    const int tid   = threadIdx.x;
    const int lane  = tid & 31;
    const int wv    = tid >> 5;          // 8 waves
    const int frame = blockIdx.x;
    const float* xg = x + (size_t)frame * (JJ * CC);
    const f32x8 fz  = {0.f,0.f,0.f,0.f,0.f,0.f,0.f,0.f};
    const u32x4 zz  = {0u,0u,0u,0u};

    // ---- stage 0: x -> bf16 LDS tile; zero padding rows 17..31; zero LN accumulators ----
    if (tid < 34) sRED[tid] = 0.f;
    for (int i = tid; i < JJ * CC / 4; i += 256) {
        const float4 v = ((const float4*)xg)[i];
        const int e = i << 2, m = e >> 9, c = e & 511;
        unsigned short* p = sXB + m * 520 + c;
        p[0] = f2bf(v.x); p[1] = f2bf(v.y); p[2] = f2bf(v.z); p[3] = f2bf(v.w);
    }
    {   // rows 17..31 of the 32-row A tile = exact zeros (no per-iter selects needed)
        u32x4* z = (u32x4*)(sXB + JJ * 520);
        for (int i = tid; i < (32 - JJ) * 520 / 8; i += 256) z[i] = zz;
    }
    __syncthreads();

    // ---- stage 1: qkv = x @ Wqkv + b  (M=17 pad 32, wave owns 192 of N=1536, 3 passes of 64) ----
    for (int pass = 0; pass < 3; ++pass) {
        const int nbase = wv * 192 + pass * 64;
        f32x8 acc[2][4];
        #pragma unroll
        for (int mt = 0; mt < 2; ++mt)
            #pragma unroll
            for (int t = 0; t < 4; ++t) acc[mt][t] = fz;

        for (int ko = 0; ko < CC; ko += 32) {
            __builtin_prefetch(WqkvT + (size_t)nbase * CC + ko + 32, 0, 1);
            const Frag a0 = fragA_fast(sXB, 520, 0,  ko);
            const Frag a1 = fragA_fast(sXB, 520, 16, ko);
            #pragma unroll
            for (int t = 0; t < 4; ++t) {
                const Frag b = fragB_glb(WqkvT, CC, nbase + t * 16, ko);
                acc[0][t] = wmma_bf16(a0, b, acc[0][t]);
                acc[1][t] = wmma_bf16(a1, b, acc[1][t]);
            }
        }
        // scatter: q[h][j][d], k[h][j][d], vT[h][d][j] (v rows j>=17 written as 0)
        #pragma unroll
        for (int mt = 0; mt < 2; ++mt)
            #pragma unroll
            for (int t = 0; t < 4; ++t) {
                const int n   = nbase + t * 16 + (lane & 15);
                const int sel = n >> 9, hh = (n >> 6) & 7, dd = n & 63;
                const float bias = bqkv[n];
                #pragma unroll
                for (int r = 0; r < 8; ++r) {
                    const int m = mt * 16 + r + ((lane >> 4) << 3);
                    const float val = (m < JJ) ? acc[mt][t][r] + bias : 0.f;
                    const unsigned short bb = f2bf(val);
                    if (sel == 2) {
                        sVT[hh * (DDIM * 40) + dd * 40 + m] = bb;
                    } else if (m < JJ) {
                        unsigned short* dst = (sel == 0) ? sQ : sK;
                        dst[hh * (JJ * 72) + m * 72 + dd] = bb;
                    }
                }
            }
    }
    __syncthreads();

    // ---- stage 2: per-head attention, wave h == head h ----
    {
        const unsigned short* sQh = sQ + wv * (JJ * 72);
        const unsigned short* sKh = sK + wv * (JJ * 72);
        f32x8 at[2][2];
        at[0][0] = fz; at[0][1] = fz; at[1][0] = fz; at[1][1] = fz;
        #pragma unroll
        for (int ko = 0; ko < DDIM; ko += 32) {
            const Frag qa0 = fragA_lds(sQh, 72, 0,  JJ, ko);
            const Frag qa1 = fragA_lds(sQh, 72, 16, JJ, ko);
            const Frag kb0 = fragB_lds(sKh, 72, 0,  JJ, ko);
            const Frag kb1 = fragB_lds(sKh, 72, 16, JJ, ko);
            at[0][0] = wmma_bf16(qa0, kb0, at[0][0]);
            at[0][1] = wmma_bf16(qa0, kb1, at[0][1]);
            at[1][0] = wmma_bf16(qa1, kb0, at[1][0]);
            at[1][1] = wmma_bf16(qa1, kb1, at[1][1]);
        }
        float* A = sAT + wv * (JJ * 20);
        #pragma unroll
        for (int mt = 0; mt < 2; ++mt)
            #pragma unroll
            for (int nt = 0; nt < 2; ++nt)
                #pragma unroll
                for (int r = 0; r < 8; ++r) {
                    const int m = mt * 16 + r + ((lane >> 4) << 3);
                    const int n = nt * 16 + (lane & 15);
                    if (m < JJ && n < JJ) A[m * 20 + n] = at[mt][nt][r] * 0.125f; // 1/sqrt(64)
                }
    }
    __syncthreads();

    // q/k now dead: zero the padding rows (17..31) of the xa tile that aliases them
    {
        u32x4* z = (u32x4*)(sXA + JJ * 520);
        for (int i = tid; i < (32 - JJ) * 520 / 8; i += 256) z[i] = zz;
    }
    // softmax over 17 keys, one row per lane
    if (lane < JJ) {
        float* row = sAT + wv * (JJ * 20) + lane * 20;
        float mx = row[0];
        for (int c = 1; c < JJ; ++c) mx = fmaxf(mx, row[c]);
        float s = 0.f;
        for (int c = 0; c < JJ; ++c) { const float e = __expf(row[c] - mx); row[c] = e; s += e; }
        const float inv = 1.f / s;
        for (int c = 0; c < JJ; ++c) row[c] *= inv;
    }
    __syncthreads();

    // sequential hierarchical skeleton-triplet adjustment (order matters)
    if (lane == 0) {
        float* A = sAT + wv * (JJ * 20);
        const int tp[13] = {0,1,0,4,0,7,8, 7, 8,11, 7, 8,14};
        const int tj[13] = {1,2,4,5,7,8,9, 8,11,12, 8,14,15};
        const int tc[13] = {2,3,5,6,8,9,10,11,12,13,14,15,16};
        #pragma unroll
        for (int i = 0; i < 13; ++i) {
            const float half = 0.5f * A[tj[i] * 20 + tp[i]];
            A[tj[i] * 20 + tc[i]] += half;
            A[tc[i] * 20 + tj[i]] += half;
        }
    }
    __syncthreads();

    // pack probs to bf16, zero-pad K 17..31 (padding must be exact zeros for WMMA)
    if (lane < JJ) {
        const float* A = sAT + wv * (JJ * 20);
        unsigned short* P = sAP + wv * (JJ * 40);
        #pragma unroll
        for (int c = 0; c < 32; ++c)
            P[lane * 40 + c] = (c < JJ) ? f2bf(A[lane * 20 + c]) : (unsigned short)0;
    }
    __syncthreads();

    // xa = attn @ v  (K = 17 padded to 32; vT rows 17..31 are zeros)
    {
        const unsigned short* P = sAP + wv * (JJ * 40);
        const unsigned short* V = sVT + wv * (DDIM * 40);
        const Frag pa0 = fragA_lds(P, 40, 0,  JJ, 0);
        const Frag pa1 = fragA_lds(P, 40, 16, JJ, 0);
        f32x8 xa[2][4];
        #pragma unroll
        for (int mt = 0; mt < 2; ++mt)
            #pragma unroll
            for (int t = 0; t < 4; ++t) xa[mt][t] = fz;
        #pragma unroll
        for (int t = 0; t < 4; ++t) {
            const Frag vb = fragB_lds(V, 40, t * 16, DDIM, 0);
            xa[0][t] = wmma_bf16(pa0, vb, xa[0][t]);
            xa[1][t] = wmma_bf16(pa1, vb, xa[1][t]);
        }
        #pragma unroll
        for (int mt = 0; mt < 2; ++mt)
            #pragma unroll
            for (int t = 0; t < 4; ++t) {
                const int n = wv * DDIM + t * 16 + (lane & 15);
                #pragma unroll
                for (int r = 0; r < 8; ++r) {
                    const int m = mt * 16 + r + ((lane >> 4) << 3);
                    if (m < JJ) sXA[m * 520 + n] = f2bf(xa[mt][t][r]);
                }
            }
    }
    __syncthreads();

    // ---- stage 3: proj GEMM + bias + residual + LayerNorm ----
    {
        const int nb = wv * 64;
        f32x8 pc[2][4];
        #pragma unroll
        for (int mt = 0; mt < 2; ++mt)
            #pragma unroll
            for (int t = 0; t < 4; ++t) pc[mt][t] = fz;

        for (int ko = 0; ko < CC; ko += 32) {
            __builtin_prefetch(WprojT + (size_t)nb * CC + ko + 32, 0, 1);
            const Frag a0 = fragA_fast(sXA, 520, 0,  ko);
            const Frag a1 = fragA_fast(sXA, 520, 16, ko);
            #pragma unroll
            for (int t = 0; t < 4; ++t) {
                const Frag b = fragB_glb(WprojT, CC, nb + t * 16, ko);
                pc[0][t] = wmma_bf16(a0, b, pc[0][t]);
                pc[1][t] = wmma_bf16(a1, b, pc[1][t]);
            }
        }

        // y = x + proj + bias; accumulate LN statistics with LDS float atomics
        #pragma unroll
        for (int mt = 0; mt < 2; ++mt)
            #pragma unroll
            for (int t = 0; t < 4; ++t) {
                const int n = nb + t * 16 + (lane & 15);
                const float bias = bproj[n];
                #pragma unroll
                for (int r = 0; r < 8; ++r) {
                    const int m = mt * 16 + r + ((lane >> 4) << 3);
                    if (m < JJ) {
                        const float y = pc[mt][t][r] + bias + bf2f(sXB[m * 520 + n]);
                        pc[mt][t][r] = y;
                        atomicAdd(&sRED[m * 2 + 0], y);
                        atomicAdd(&sRED[m * 2 + 1], y * y);
                    }
                }
            }
        __syncthreads();

        float* og = out + (size_t)frame * (JJ * CC);
        #pragma unroll
        for (int mt = 0; mt < 2; ++mt)
            #pragma unroll
            for (int t = 0; t < 4; ++t) {
                const int n = nb + t * 16 + (lane & 15);
                #pragma unroll
                for (int r = 0; r < 8; ++r) {
                    const int m = mt * 16 + r + ((lane >> 4) << 3);
                    if (m < JJ) {
                        const float mu  = sRED[m * 2 + 0] * (1.f / 512.f);
                        const float var = sRED[m * 2 + 1] * (1.f / 512.f) - mu * mu;
                        og[m * CC + n] = (pc[mt][t][r] - mu) * rsqrtf(var + 1e-5f);
                    }
                }
            }
    }
}

extern "C" void kernel_launch(void* const* d_in, const int* in_sizes, int n_in,
                              void* d_out, int out_size, void* d_ws, size_t ws_size,
                              hipStream_t stream) {
    const float* x     = (const float*)d_in[0];
    const float* Wqkv  = (const float*)d_in[1];
    const float* bqkv  = (const float*)d_in[2];
    const float* Wproj = (const float*)d_in[3];
    const float* bproj = (const float*)d_in[4];

    unsigned short* WqkvT  = (unsigned short*)d_ws;            // 1536*512 bf16
    unsigned short* WprojT = WqkvT + (size_t)CC * NC3;         //  512*512 bf16

    cvt_weights<<<(CC * NC3 + 255) / 256, 256, 0, stream>>>(Wqkv, Wproj, WqkvT, WprojT);
    hsdm_fused<<<BN, 256, SMEM_BYTES, stream>>>(x, bqkv, bproj, WqkvT, WprojT, (float*)d_out);
}
